// unnamed_model3_58506044506599
// MI455X (gfx1250) — matrix operand order repaired and checked by reading, not compiled, not run on hardware
//
#include <hip/hip_runtime.h>

// ---------------------------------------------------------------------------
// GCN 2-layer forward for MI455X (gfx1250, wave32, WMMA + async-to-LDS)
// ---------------------------------------------------------------------------

typedef __bf16 bf16_t;
typedef bf16_t v8bf  __attribute__((ext_vector_type(8)));
typedef bf16_t v16bf __attribute__((ext_vector_type(16)));
typedef float  v8f   __attribute__((ext_vector_type(8)));

#define NFEAT 128

// ---------------------------------------------------------------------------
// Elementwise float -> bf16 convert (optionally fused ReLU)
// ---------------------------------------------------------------------------
__global__ void cvt_f32_bf16(const float* __restrict__ src, bf16_t* __restrict__ dst,
                             int n, int relu) {
    int i = blockIdx.x * blockDim.x + threadIdx.x;
    if (i >= n) return;
    float v = src[i];
    if (relu) v = fmaxf(v, 0.0f);
    dst[i] = (bf16_t)v;
}

// Weight convert + transpose: src K x N (row-major) -> dst N x K (bf16)
__global__ void cvt_transpose_bf16(const float* __restrict__ src, bf16_t* __restrict__ dst,
                                   int K, int N) {
    int idx = blockIdx.x * blockDim.x + threadIdx.x;
    if (idx >= K * N) return;
    int k = idx / N, n = idx - k * N;
    dst[(size_t)n * K + k] = (bf16_t)src[idx];
}

// ---------------------------------------------------------------------------
// Degree / normalization setup
// ---------------------------------------------------------------------------
__global__ void deg_init(float* __restrict__ deg, int n) {
    int i = blockIdx.x * blockDim.x + threadIdx.x;
    if (i < n) deg[i] = 1.0f;   // self-loop weight
}

__global__ void deg_accum(const int* __restrict__ ei, const float* __restrict__ ew,
                          float* __restrict__ deg, int E) {
    int e = blockIdx.x * blockDim.x + threadIdx.x;
    if (e >= E) return;
    atomicAdd(&deg[ei[E + e]], ew[e]);
}

__global__ void deg_to_dinv(float* __restrict__ deg, int n) {
    int i = blockIdx.x * blockDim.x + threadIdx.x;
    if (i >= n) return;
    float d = deg[i];
    deg[i] = (d > 0.0f) ? rsqrtf(d) : 0.0f;
}

__global__ void edge_norm(const int* __restrict__ ei, const float* __restrict__ ew,
                          const float* __restrict__ dinv, float* __restrict__ norm,
                          int E) {
    int e = blockIdx.x * blockDim.x + threadIdx.x;
    if (e >= E) return;
    norm[e] = dinv[ei[e]] * ew[e] * dinv[ei[E + e]];
}

// ---------------------------------------------------------------------------
// Async global -> LDS copy helper (gfx1250 ASYNC path, 16B per lane-op)
// count16 = number of 16-byte chunks; all 256 threads participate.
// ---------------------------------------------------------------------------
__device__ __forceinline__ void async_copy_to_lds(bf16_t* lds_dst, const bf16_t* g_src,
                                                  int count16, int tid) {
    for (int idx = tid; idx < count16; idx += 256) {
        unsigned lds = (unsigned)(size_t)(lds_dst + idx * 8);   // LDS byte offset (low 32b)
        const bf16_t* g = g_src + idx * 8;
        asm volatile("global_load_async_to_lds_b128 %0, %1, off"
                     :: "v"(lds), "v"(g) : "memory");
    }
    asm volatile("s_wait_asynccnt 0x0" ::: "memory");
}

// ---------------------------------------------------------------------------
// WMMA GEMM:  C[M x NOUT] = A[M x K] * Bt^T  (Bt is NOUT x K transposed weights)
// Block = 256 threads = 8 waves, covers MT*16 rows x NOUT cols.
// Wave w owns column tile [16w, 16w+16) and iterates MT row tiles.
// A/B fragments hold k = 16*(lane/16) + s, contiguous (wave32); all LDS reads are b128.
// ---------------------------------------------------------------------------
template <int K, int NOUT, int MT>
__global__ void gemm_wmma_bf16(const bf16_t* __restrict__ A,
                               const bf16_t* __restrict__ Bt,   // NOUT x K
                               float* __restrict__ C, int M) {
    __shared__ bf16_t Alds[MT * 16 * K];    // 20 KB for MT=5, K=128
    __shared__ bf16_t Blds[NOUT * K];       // <= 32 KB

    const int tid     = threadIdx.x;
    const int wave    = tid >> 5;
    const int lane    = tid & 31;
    const int rowBase = blockIdx.x * (MT * 16);

    // Stage A tile (MT*16 x K) and Bt (NOUT x K) via async-to-LDS (bf16, linear)
    const int rowsHere = min(MT * 16, M - rowBase);
    async_copy_to_lds(Alds, A + (size_t)rowBase * K, rowsHere * K / 8, tid);
    async_copy_to_lds(Blds, Bt, NOUT * K / 8, tid);
    __syncthreads();

    constexpr int NTILES = NOUT / 16;
    if (wave < NTILES) {
        const int nbase   = wave << 4;
        const int halfsel = lane >> 4;     // 0: lanes 0-15, 1: lanes 16-31
        const int lm      = lane & 15;

        v8f acc[MT];
        #pragma unroll
        for (int mt = 0; mt < MT; ++mt) acc[mt] = (v8f){};

        #pragma unroll
        for (int kc = 0; kc < K; kc += 32) {
            // B fragment: N = nbase+lm, K-run contiguous in transposed layout
            const bf16_t* bp = &Blds[(nbase + lm) * K + kc + halfsel * 16];
            v8bf blo = *(const v8bf*)(bp);
            v8bf bhi = *(const v8bf*)(bp + 8);
            v16bf b = __builtin_shufflevector(blo, bhi,
                0,1,2,3,4,5,6,7,8,9,10,11,12,13,14,15);

            #pragma unroll
            for (int mt = 0; mt < MT; ++mt) {
                // A fragment: row = mt*16+lm, one contiguous 16-element K run
                const bf16_t* ap = &Alds[(mt * 16 + lm) * K + kc + halfsel * 16];
                v8bf alo = *(const v8bf*)(ap);
                v8bf ahi = *(const v8bf*)(ap + 8);
                v16bf a = __builtin_shufflevector(alo, ahi,
                    0,1,2,3,4,5,6,7,8,9,10,11,12,13,14,15);

                acc[mt] = __builtin_amdgcn_wmma_f32_16x16x32_bf16(
                    false, a, false, b, (short)0, acc[mt], false, false);
            }
        }

        // C layout: VGPR r -> row (halfsel*8 + r), col = nbase + lm
        #pragma unroll
        for (int mt = 0; mt < MT; ++mt) {
            #pragma unroll
            for (int r = 0; r < 8; ++r) {
                int grow = rowBase + mt * 16 + halfsel * 8 + r;
                if (grow < M)
                    C[(size_t)grow * NOUT + nbase + lm] = acc[mt][r];
            }
        }
    }
}

// ---------------------------------------------------------------------------
// Aggregation init: out[i,j] = bias[j] + dinv[i]^2 * H[i,j]   (self-loop term)
// ---------------------------------------------------------------------------
__global__ void agg_init(const float* __restrict__ H, const float* __restrict__ bias,
                         const float* __restrict__ dinv, float* __restrict__ out,
                         int n, int F) {
    int idx = blockIdx.x * blockDim.x + threadIdx.x;
    if (idx >= n * F) return;
    int i = idx / F;
    int j = idx - i * F;
    float di = dinv[i];
    out[idx] = bias[j] + di * di * H[idx];
}

// ---------------------------------------------------------------------------
// Edge scatter: out[col] += norm[e] * H[row]; one wave per edge,
// lane-strided columns -> coalesced gathers + coalesced f32 atomics (L2-resident)
// ---------------------------------------------------------------------------
__global__ void scatter_edges(const int* __restrict__ ei, const float* __restrict__ norm,
                              const float* __restrict__ H, float* __restrict__ out,
                              int E, int F) {
    int e = blockIdx.x * 8 + (threadIdx.x >> 5);
    if (e >= E) return;
    int lane = threadIdx.x & 31;
    int row  = ei[e];
    int col  = ei[E + e];
    float w  = norm[e];
    const float* __restrict__ hr = H + (size_t)row * F;
    float* __restrict__ orow     = out + (size_t)col * F;
    for (int c = lane; c < F; c += 32) {
        atomicAdd(&orow[c], w * hr[c]);
    }
}

// ---------------------------------------------------------------------------
// Row-wise L2 normalize (one wave per row)
// ---------------------------------------------------------------------------
__global__ void l2norm_rows(float* __restrict__ out, int n, int F) {
    int row = blockIdx.x * 8 + (threadIdx.x >> 5);
    if (row >= n) return;
    int lane = threadIdx.x & 31;
    float* __restrict__ r = out + (size_t)row * F;
    float s = 0.0f;
    for (int c = lane; c < F; c += 32) { float v = r[c]; s += v * v; }
    #pragma unroll
    for (int off = 16; off > 0; off >>= 1) s += __shfl_xor(s, off, 32);
    float inv = 1.0f / fmaxf(sqrtf(s), 1e-12f);
    for (int c = lane; c < F; c += 32) r[c] *= inv;
}

// ---------------------------------------------------------------------------
// Launcher
// ---------------------------------------------------------------------------
static inline size_t align256(size_t x) { return (x + 255) & ~(size_t)255; }

extern "C" void kernel_launch(void* const* d_in, const int* in_sizes, int n_in,
                              void* d_out, int out_size, void* d_ws, size_t ws_size,
                              hipStream_t stream) {
    const float* x  = (const float*)d_in[0];
    const int*   ei = (const int*)  d_in[1];
    const float* ew = (const float*)d_in[2];
    const float* W1 = (const float*)d_in[3];
    const float* b1 = (const float*)d_in[4];
    const float* W2 = (const float*)d_in[5];
    const float* b2 = (const float*)d_in[6];
    float* out = (float*)d_out;

    const int N      = in_sizes[0] / NFEAT;      // 50000
    const int E      = in_sizes[2];              // 800000
    const int NHID   = in_sizes[4];              // 128
    const int NCLASS = in_sizes[6];              // 64

    // ---- workspace carve-out (with reuse) ----
    char* base = (char*)d_ws;
    size_t off = 0;
    float*  dinv  = (float*)(base + off);  off += align256((size_t)N * 4);
    float*  norm  = (float*)(base + off);  off += align256((size_t)E * 4);
    bf16_t* Xbf   = (bf16_t*)(base + off); off += align256((size_t)N * NFEAT * 2); // reused as H1bf
    bf16_t* W1t   = (bf16_t*)(base + off); off += align256((size_t)NFEAT * NHID * 2);
    bf16_t* W2t   = (bf16_t*)(base + off); off += align256((size_t)NHID * NCLASS * 2);
    float*  HX    = (float*)(base + off);  off += align256((size_t)N * NHID * 4);  // reused as HY
    float*  AGG1  = (float*)(base + off);  off += align256((size_t)N * NHID * 4);
    bf16_t* H1bf  = Xbf;   // X no longer needed after GEMM1
    float*  HY    = HX;    // HX no longer needed after layer-1 aggregation
    (void)ws_size; (void)n_in; (void)out_size;

    const int T = 256;
    #define CDIV(a,b) (((a)+(b)-1)/(b))

    // 1) bf16 conversions (weights transposed to N x K for contiguous B frags)
    cvt_f32_bf16      <<<CDIV(N * NFEAT, T), T, 0, stream>>>(x, Xbf, N * NFEAT, 0);
    cvt_transpose_bf16<<<CDIV(NFEAT * NHID, T), T, 0, stream>>>(W1, W1t, NFEAT, NHID);
    cvt_transpose_bf16<<<CDIV(NHID * NCLASS, T), T, 0, stream>>>(W2, W2t, NHID, NCLASS);

    // 2) degree -> dinv -> per-edge norm
    deg_init   <<<CDIV(N, T), T, 0, stream>>>(dinv, N);
    deg_accum  <<<CDIV(E, T), T, 0, stream>>>(ei, ew, dinv, E);
    deg_to_dinv<<<CDIV(N, T), T, 0, stream>>>(dinv, N);
    edge_norm  <<<CDIV(E, T), T, 0, stream>>>(ei, ew, dinv, norm, E);

    // 3) layer 1: WMMA GEMM -> self-loop init -> edge scatter -> ReLU+cvt
    gemm_wmma_bf16<128, 128, 5><<<CDIV(N, 80), T, 0, stream>>>(Xbf, W1t, HX, N);
    agg_init      <<<CDIV(N * NHID, T), T, 0, stream>>>(HX, b1, dinv, AGG1, N, NHID);
    scatter_edges <<<CDIV(E, 8), T, 0, stream>>>(ei, norm, HX, AGG1, E, NHID);
    cvt_f32_bf16  <<<CDIV(N * NHID, T), T, 0, stream>>>(AGG1, H1bf, N * NHID, 1);

    // 4) layer 2: WMMA GEMM -> self-loop init -> edge scatter
    gemm_wmma_bf16<128, 64, 5><<<CDIV(N, 80), T, 0, stream>>>(H1bf, W2t, HY, N);
    agg_init      <<<CDIV(N * NCLASS, T), T, 0, stream>>>(HY, b2, dinv, out, N, NCLASS);
    scatter_edges <<<CDIV(E, 8), T, 0, stream>>>(ei, norm, HY, out, E, NCLASS);

    // 5) final row L2 normalize
    l2norm_rows<<<CDIV(N, 8), T, 0, stream>>>(out, N, NCLASS);
}
